// TTSModel_45432164057636
// MI455X (gfx1250) — matrix operand order repaired and checked by reading, not compiled, not run on hardware
//
#include <hip/hip_runtime.h>
#include <hip/hip_bf16.h>

#define B_   16
#define NMEL 80
#define TX   512
#define TY   2048
#define NEGV (-1e9f)

typedef __attribute__((ext_vector_type(16))) _Float16 v16h;
typedef __attribute__((ext_vector_type(8)))  float    v8f;

// ---------------------------------------------------------------------------
// Kernel 1: rowterm[b,t] = -0.5*sum_c x_mu^2 ; colterm[b,f] = -0.5*sum_c z^2
// ---------------------------------------------------------------------------
__global__ void glowtts_stats(const float* __restrict__ xmu,
                              const float* __restrict__ z,
                              float* __restrict__ rowterm,
                              float* __restrict__ colterm) {
  int i = blockIdx.x * blockDim.x + threadIdx.x;
  if (i < B_ * TX) {
    int b = i / TX, t = i % TX;
    const float* p = xmu + (size_t)b * NMEL * TX + t;
    float s = 0.f;
#pragma unroll 8
    for (int c = 0; c < NMEL; ++c) { float v = p[(size_t)c * TX]; s += v * v; }
    rowterm[i] = -0.5f * s;
  } else if (i < B_ * TX + B_ * TY) {
    int j = i - B_ * TX;
    int b = j / TY, f = j % TY;
    const float* p = z + (size_t)b * NMEL * TY + f;
    float s = 0.f;
#pragma unroll 8
    for (int c = 0; c < NMEL; ++c) { float v = p[(size_t)c * TY]; s += v * v; }
    colterm[j] = -0.5f * s;
  }
}

// ---------------------------------------------------------------------------
// Kernel 2: logp = A(16x80 tile of x_mu^T) @ B(80x64 tile of z) via
// v_wmma_f32_16x16x32_f16, epilogue adds rowterm+colterm+C0.
// Each wave computes a 16(M) x 64(N) strip; K=80 padded to 96 (3 x K32).
// ---------------------------------------------------------------------------
__global__ __launch_bounds__(256) void glowtts_logp_wmma(
    const float* __restrict__ xmu, const float* __restrict__ z,
    const float* __restrict__ rowterm, const float* __restrict__ colterm,
    float* __restrict__ logp) {
  const int tid  = threadIdx.x;
  const int lane = tid & 31;
  const int wave = tid >> 5;
  const int strip = blockIdx.x * 8 + wave;   // B_*32*32 = 16384 strips
  const int b  = strip >> 10;
  const int rm = strip & 1023;
  const int t0 = (rm >> 5) * 16;             // M tile
  const int f0 = (rm & 31) * 64;             // N strip (4 subtiles)

  const int m  = lane & 15;                  // M (A) / N (B,C,D) index
  const int kh = lane >> 4;                  // K-half selector
  const int t  = t0 + m;

  v8f acc[4] = {};

  const float* Ab = xmu + (size_t)b * NMEL * TX;
  const float* Bb = z   + (size_t)b * NMEL * TY;

#pragma unroll
  for (int kc = 0; kc < 3; ++kc) {
    const int k0 = kc * 32;
    // A fragment: 16-bit A 16x32 layout -> lane m=M, K(j) = kh*8 + (j&7) + (j>>3)*16
    v16h a;
#pragma unroll
    for (int j = 0; j < 16; ++j) {
      int K = k0 + kh * 8 + (j & 7) + ((j >> 3) << 4);
      a[j] = (K < NMEL) ? (_Float16)Ab[(size_t)K * TX + t] : (_Float16)0.0f;
    }
#pragma unroll
    for (int s = 0; s < 4; ++s) {
      const int f = f0 + s * 16 + m;
      // B fragment: 16-bit B 32x16 layout -> lane m=N, K(j) = kh*8 + (j&7) + (j>>3)*16
      v16h bm;
#pragma unroll
      for (int j = 0; j < 16; ++j) {
        int K = k0 + (kh << 3) + (j & 7) + ((j >> 3) << 4);
        bm[j] = (K < NMEL) ? (_Float16)Bb[(size_t)K * TY + f] : (_Float16)0.0f;
      }
      acc[s] = __builtin_amdgcn_wmma_f32_16x16x32_f16(
          /*neg_a=*/false, a, /*neg_b=*/false, bm,
          /*c_mod=*/(short)0, acc[s], /*reuse_a=*/false, /*reuse_b=*/false);
    }
  }

  // Epilogue. D layout: VGPR rr -> M = kh*8 + rr, N = lane&15.
  const float C0 = -0.91893853320467274178f * (float)NMEL;
  float rt[8];
#pragma unroll
  for (int rr = 0; rr < 8; ++rr)
    rt[rr] = rowterm[b * TX + t0 + (kh << 3) + rr];
#pragma unroll
  for (int s = 0; s < 4; ++s) {
    const int f  = f0 + s * 16 + m;
    const float ct = colterm[b * TY + f];
#pragma unroll
    for (int rr = 0; rr < 8; ++rr) {
      const int trow = t0 + (kh << 3) + rr;
      logp[((size_t)b * TX + trow) * TY + f] = acc[s][rr] + rt[rr] + ct + C0;
    }
  }
}

// ---------------------------------------------------------------------------
// Kernel 3: monotonic alignment search. One 512-thread block per batch.
// Forward: value column double-buffered in LDS, logp staged in LDS tile
// (pad 17 -> conflict-free column reads), decisions recorded via wave32
// ballot into ws bits. Backward: bits chunk-staged in LDS, thread 0 walks.
// ---------------------------------------------------------------------------
__global__ __launch_bounds__(512) void glowtts_mas(
    const float* __restrict__ logp, const int* __restrict__ xlen_,
    const int* __restrict__ ylen_, unsigned* __restrict__ bits,
    float* __restrict__ path, float* __restrict__ dur_out) {
  __shared__ float    tile[TX][17];    // 512 x 16 y-chunk, padded
  __shared__ float    vbuf[2][TX];
  __shared__ unsigned bb[128 * 16];    // backward staging: 128 y x 16 words
  __shared__ int      dur[TX];

  const int b    = blockIdx.x;
  const int tid  = threadIdx.x;
  const int x    = tid;
  const int xlen = xlen_[b];
  const int ylen = ylen_[b];

  // zero this batch's path slice
  float4* pz = (float4*)(path + (size_t)b * TX * TY);
  for (int i = tid; i < TX * TY / 4; i += 512) pz[i] = make_float4(0, 0, 0, 0);
  dur[tid]   = 0;
  vbuf[0][x] = NEGV;
  __syncthreads();

  // ---- forward DP ----
  int p = 0;
  const float* lp = logp + (size_t)b * TX * TY;
  const bool   xv = (x < xlen);
  for (int yc = 0; yc < TY / 16; ++yc) {
    // coalesced tile load: 16 lanes read 16 consecutive y per row
#pragma unroll 1
    for (int i = 0; i < 16; ++i) {
      int lrow = i * 32 + (tid >> 4);
      int yj   = tid & 15;
      tile[lrow][yj] = lp[(size_t)lrow * TY + yc * 16 + yj];
    }
    __syncthreads();
    for (int j = 0; j < 16; ++j) {
      const int   y    = yc * 16 + j;
      const float vx   = vbuf[p][x];
      const float vxm1 = (x > 0) ? vbuf[p][x - 1] : NEGV;
      const float vc   = (x == y) ? NEGV : vx;
      const float vpv  = (x == 0) ? ((y == 0) ? 0.0f : NEGV) : vxm1;
      const float rlp  = (xv && (y < ylen)) ? tile[x][j] : NEGV;
      vbuf[p ^ 1][x]   = fmaxf(vc, vpv) + rlp;
      // backtrack decision: (x==y) || vals[x,y-1] < vals[x-1,y-1]
      const bool c = (x == y) || (vx < vxm1);
      unsigned msk = __builtin_amdgcn_ballot_w32(c);
      if ((tid & 31) == 0) bits[((size_t)b * TY + y) * 16 + (tid >> 5)] = msk;
      __syncthreads();
      p ^= 1;
    }
  }

  // ---- backward walk ----
  __threadfence();
  __syncthreads();
  int idx = xlen - 1;
  for (int yc = TY / 128 - 1; yc >= 0; --yc) {
    for (int i = tid; i < 128 * 16; i += 512)
      bb[i] = bits[((size_t)b * TY + yc * 128) * 16 + i];
    __syncthreads();
    if (tid == 0) {
      for (int j = 127; j >= 0; --j) {
        const int y = yc * 128 + j;
        if (y < ylen) {
          path[((size_t)b * TX + idx) * TY + y] = 1.0f;
          dur[idx]++;
          const unsigned w = bb[j * 16 + (idx >> 5)];
          if (idx > 0 && ((w >> (idx & 31)) & 1u)) idx--;
        }
      }
    }
    __syncthreads();
  }
  dur_out[b * TX + x] = (float)dur[x];
}

// ---------------------------------------------------------------------------
extern "C" void kernel_launch(void* const* d_in, const int* in_sizes, int n_in,
                              void* d_out, int out_size, void* d_ws,
                              size_t ws_size, hipStream_t stream) {
  const float* xmu  = (const float*)d_in[0];
  const float* z    = (const float*)d_in[1];
  const int*   xlen = (const int*)d_in[2];
  const int*   ylen = (const int*)d_in[3];

  float* logp = (float*)d_out;
  float* path = logp + (size_t)B_ * TX * TY;
  float* dur  = path + (size_t)B_ * TX * TY;

  float*    rowterm = (float*)d_ws;                 //  8192 f32
  float*    colterm = rowterm + B_ * TX;            // 32768 f32
  unsigned* bits    = (unsigned*)(colterm + B_ * TY); // 16*2048*16 u32 (2 MB)

  glowtts_stats<<<(B_ * TX + B_ * TY + 255) / 256, 256, 0, stream>>>(
      xmu, z, rowterm, colterm);
  glowtts_logp_wmma<<<(B_ * 32 * 32) / 8, 256, 0, stream>>>(
      xmu, z, rowterm, colterm, logp);
  glowtts_mas<<<B_, 512, 0, stream>>>(logp, xlen, ylen, bits, path, dur);
}
